// MultiheadPerformerAttention_89678917140633
// MI455X (gfx1250) — matrix operand order repaired and checked by reading, not compiled, not run on hardware
//
#include <hip/hip_runtime.h>
#include <hip/hip_bf16.h>

// ---------------- types ----------------
typedef __attribute__((ext_vector_type(16))) __bf16 v16bf;
typedef __attribute__((ext_vector_type(8)))  __bf16 v8bf;
typedef __attribute__((ext_vector_type(8)))  float  v8f;
typedef unsigned int uint32x4 __attribute__((ext_vector_type(4)));
typedef int          int32x8  __attribute__((ext_vector_type(8)));
typedef int          int32x4  __attribute__((ext_vector_type(4)));

#define WMMA_BF16(A, B, C) \
  __builtin_amdgcn_wmma_f32_16x16x32_bf16(false, (A), false, (B), (short)0, (C), false, false)

// Problem constants
#define NSEQ 4096
#define BB   4
#define EE   1024
#define HH   16
#define DD   64
#define MM   64
#define NB   (NSEQ*BB)   // 16384
#define BH   (BB*HH)     // 64

// ---------------- fragment loaders ----------------
// A fragment: 16x32 bf16, row m = lane&15, half = lane>>4 covers K chunk
// [half*16, half*16+16) (contiguous, matches B swizzle)
__device__ inline v16bf load_a_frag(const __bf16* tile, long ld) {
  int lane = threadIdx.x & 31;
  int m = lane & 15, h = lane >> 4;
  const __bf16* row = tile + (long)m * ld;
  return *(const v16bf*)(row + h * 16);
}

// B fragment: pre-swizzled blob of 512 bf16; lane reads 16 contiguous values.
// swizzle: value j of lane L = B[k = kt*32 + (L>>4)*16 + j][n = nt*16 + (L&15)]
__device__ inline v16bf load_b_frag(const __bf16* frag) {
  int lane = threadIdx.x & 31;
  return *(const v16bf*)(frag + lane * 16);
}

// ---------------- Tensor Data Mover: 2D tile (8B elements) global -> LDS ----------------
// D# layout per CDNA5 ISA ch.8: group0 = count/lds_addr/global_addr/type,
// group1 = data_size, tensor dims, tile dims, dim0 stride. groups 2,3 = 0 (2D).
__device__ inline void tdm_load_2d(unsigned lds_addr, const void* gaddr,
                                   unsigned tile0_elems, unsigned tile1_rows,
                                   unsigned long long row_stride_elems) {
  unsigned long long ga = (unsigned long long)gaddr;
  uint32x4 g0;
  g0[0] = 1u;                                                 // count = 1 valid descriptor
  g0[1] = lds_addr;                                           // LDS byte address
  g0[2] = (unsigned)(ga & 0xffffffffu);                       // global_addr[31:0]
  g0[3] = (unsigned)((ga >> 32) & 0x1ffffffu) | (2u << 30);   // global_addr[56:32] | type=2
  const unsigned long long td0 = 0x40000000ull;               // huge tensor dims: no OOB clip
  const unsigned long long td1 = 0x40000000ull;
  unsigned long long s0 = row_stride_elems;                   // dim0 stride (8B units)
  int32x8 g1;
  g1[0] = (int)(3u << 16);                                    // data_size = 3 (8 bytes)
  g1[1] = (int)((td0 & 0xffffu) << 16);                       // tensor_dim0[15:0] @ bits 63:48
  g1[2] = (int)(((td0 >> 16) & 0xffffu) | ((td1 & 0xffffu) << 16));
  g1[3] = (int)(((td1 >> 16) & 0xffffu) | ((tile0_elems & 0xffffu) << 16));
  g1[4] = (int)(tile1_rows & 0xffffu);                        // tile_dim1, tile_dim2=0
  g1[5] = (int)(s0 & 0xffffffffu);
  g1[6] = (int)((s0 >> 32) & 0xffffu);                        // stride0[47:32], stride1=0
  g1[7] = 0;
  int32x4 z4 = {0, 0, 0, 0};
  int32x8 z8 = {0, 0, 0, 0, 0, 0, 0, 0};
  __builtin_amdgcn_tensor_load_to_lds(g0, g1, z4, z4, z8, 0);
}

// ---------------- swizzle kernel (builds B fragments) ----------------
template <typename SrcT>
__global__ void swizzleB(const SrcT* __restrict__ src, __bf16* __restrict__ dst,
                         long baseStride, long strideK, long strideN,
                         int Ntiles, long fragsPerZ, float scale) {
  int frag = blockIdx.x * (blockDim.x >> 5) + (threadIdx.x >> 5);
  int lane = threadIdx.x & 31;
  int z = blockIdx.y;
  if (frag >= fragsPerZ) return;
  int kt = frag / Ntiles, nt = frag % Ntiles;
  long srcBase = (long)z * baseStride;
  long n = nt * 16 + (lane & 15);
  __bf16* d = dst + ((long)z * fragsPerZ + frag) * 512 + lane * 16;
#pragma unroll
  for (int j = 0; j < 16; ++j) {
    long k = kt * 32 + (lane >> 4) * 16 + j;
    d[j] = (__bf16)((float)src[srcBase + k * strideK + n * strideN] * scale);
  }
}

// ---------------- f32 -> bf16 ----------------
__global__ void cvt_f32_bf16(const float* __restrict__ s, __bf16* __restrict__ d, long n) {
  long i = (long)blockIdx.x * blockDim.x + threadIdx.x;
  if (i < n) d[i] = (__bf16)s[i];
}

// ---------------- generic WMMA GEMM with TDM-staged B operand ----------------
// Y[r,c] = X[r,:].Bsw + bias[c]. Block = 8 waves; each wave owns a 32x64 tile
// (8 accumulators, 2 A fragments sharing 4 B fragments per k-step).
// B strip (shared by all 8 waves) is DMA'd into LDS by the Tensor Data Mover,
// double-buffered in GCHUNK k-step slabs, WMMAs read it back via ds_load.
#define GCHUNK 8
__global__ void gemm_bias(const __bf16* __restrict__ X, const __bf16* __restrict__ Bsw,
                          const float* __restrict__ bias,
                          __bf16* __restrict__ Ybf, float* __restrict__ Yf,
                          int Kt, int NtTot, long ldx, long ldy) {
  __shared__ __bf16 bsm[2][GCHUNK * 4 * 512];   // 2 x 32 KB
  int wave = threadIdx.x >> 5, lane = threadIdx.x & 31;
  long row0 = ((long)blockIdx.x * (blockDim.x >> 5) + wave) * 32;
  int nt0 = blockIdx.y * 4;
  const int NCH = Kt / GCHUNK;
  v8f acc[8] = {};

  // chunk c covers kt in [c*GCHUNK, (c+1)*GCHUNK); fragment (kt, nt0+j) lives at
  // (kt*NtTot + nt0 + j)*512 elems -> per kt a contiguous 4 KB row, row stride NtTot*1 KB.
  if (threadIdx.x == 0)
    tdm_load_2d((unsigned)(unsigned long long)(const void*)&bsm[0][0],
                Bsw + ((long)0 * NtTot + nt0) * 512,
                /*tile0(8B elems)*/ 4 * 512 / 4, /*rows*/ GCHUNK,
                /*row stride(8B elems)*/ (unsigned long long)NtTot * 512 / 4);

  for (int c = 0; c < NCH; ++c) {
    if (threadIdx.x == 0) {
      if (c + 1 < NCH) {
        tdm_load_2d((unsigned)(unsigned long long)(const void*)&bsm[(c + 1) & 1][0],
                    Bsw + ((long)(c + 1) * GCHUNK * NtTot + nt0) * 512,
                    4 * 512 / 4, GCHUNK, (unsigned long long)NtTot * 512 / 4);
        __builtin_amdgcn_s_wait_tensorcnt(1);   // chunk c complete (in-order)
      } else {
        __builtin_amdgcn_s_wait_tensorcnt(0);   // last chunk complete
      }
    }
    __syncthreads();   // chunk c resident in bsm[c&1]
    const __bf16* bb = &bsm[c & 1][0];
#pragma unroll
    for (int y = 0; y < GCHUNK; ++y) {
      int kt = c * GCHUNK + y;
      __builtin_prefetch((const void*)(X + row0 * ldx + (long)(kt + 2) * 32), 0, 3);
      // load everything first so a single wait covers the whole WMMA group
      v16bf b0 = load_b_frag(bb + (y * 4 + 0) * 512);
      v16bf b1 = load_b_frag(bb + (y * 4 + 1) * 512);
      v16bf b2 = load_b_frag(bb + (y * 4 + 2) * 512);
      v16bf b3 = load_b_frag(bb + (y * 4 + 3) * 512);
      v16bf aL = load_a_frag(X + row0 * ldx + (long)kt * 32, ldx);
      v16bf aH = load_a_frag(X + (row0 + 16) * ldx + (long)kt * 32, ldx);
      acc[0] = WMMA_BF16(aL, b0, acc[0]);
      acc[1] = WMMA_BF16(aL, b1, acc[1]);
      acc[2] = WMMA_BF16(aL, b2, acc[2]);
      acc[3] = WMMA_BF16(aL, b3, acc[3]);
      acc[4] = WMMA_BF16(aH, b0, acc[4]);
      acc[5] = WMMA_BF16(aH, b1, acc[5]);
      acc[6] = WMMA_BF16(aH, b2, acc[6]);
      acc[7] = WMMA_BF16(aH, b3, acc[7]);
    }
    __syncthreads();   // all waves done reading before TDM re-targets this buffer
  }

  int half = lane >> 4, n16 = lane & 15;
#pragma unroll
  for (int g = 0; g < 2; ++g) {
#pragma unroll
    for (int j = 0; j < 4; ++j) {
      long col = (long)(nt0 + j) * 16 + n16;
      float bc = bias ? bias[col] : 0.f;
#pragma unroll
      for (int v = 0; v < 8; ++v) {
        long r = row0 + g * 16 + half * 8 + v;
        float y = acc[g * 4 + j][v] + bc;
        if (Ybf) Ybf[r * ldy + col] = (__bf16)y;
        if (Yf)  Yf[r * ldy + col] = y;
      }
    }
  }
}

// ---------------- FAVOR+ helpers ----------------
__device__ inline void ddash_tile(const __bf16* abase, long lda, const __bf16* frags,
                                  v8f acc[4], v16bf& a0, v16bf& a1) {
  a0 = load_a_frag(abase, lda);
  a1 = load_a_frag(abase + 32, lda);
  v16bf b0 = load_b_frag(frags + 0 * 512);
  v16bf b1 = load_b_frag(frags + 1 * 512);
  v16bf b2 = load_b_frag(frags + 2 * 512);
  v16bf b3 = load_b_frag(frags + 3 * 512);
  acc[0] = WMMA_BF16(a0, b0, acc[0]);
  acc[1] = WMMA_BF16(a0, b1, acc[1]);
  acc[2] = WMMA_BF16(a0, b2, acc[2]);
  acc[3] = WMMA_BF16(a0, b3, acc[3]);
  b0 = load_b_frag(frags + 4 * 512);
  b1 = load_b_frag(frags + 5 * 512);
  b2 = load_b_frag(frags + 6 * 512);
  b3 = load_b_frag(frags + 7 * 512);
  acc[0] = WMMA_BF16(a1, b0, acc[0]);
  acc[1] = WMMA_BF16(a1, b1, acc[1]);
  acc[2] = WMMA_BF16(a1, b2, acc[2]);
  acc[3] = WMMA_BF16(a1, b3, acc[3]);
}

__device__ inline float rowsumsq(const v16bf& a0, const v16bf& a1) {
  float ss = 0.f;
#pragma unroll
  for (int i = 0; i < 16; ++i) {
    float x = (float)a0[i]; ss += x * x;
    float y = (float)a1[i]; ss += y * y;
  }
  ss += __shfl_xor(ss, 16);  // combine complementary K halves (same row m)
  return ss;                 // lane L holds sum of squares of row (L & 15)
}

__device__ inline void atomicMaxF(float* a, float v) {
  if (v >= 0.f) atomicMax((int*)a, __float_as_int(v));
  else          atomicMin((unsigned int*)a, __float_as_uint(v));
}

// FAVOR+ for q: per-row stabilizer. qprime[bh][n][m] bf16.
__global__ void favorp_q_kernel(const __bf16* __restrict__ Qh, const __bf16* __restrict__ projsw,
                                __bf16* __restrict__ qprime) {
  int wave = threadIdx.x >> 5, lane = threadIdx.x & 31;
  int tile = blockIdx.x * (blockDim.x >> 5) + wave;
  int bh = blockIdx.y, b = bh >> 4, h = bh & 15;
  long row0 = (long)tile * 16;
  const __bf16* abase = Qh + (row0 * BB + b) * EE + h * DD;  // row stride BB*EE
  v8f acc[4] = {}; v16bf a0, a1;
  ddash_tile(abase, (long)BB * EE, projsw + (long)h * 8 * 512, acc, a0, a1);
  float ss = rowsumsq(a0, a1);
  int half = lane >> 4, n16 = lane & 15;
  float stab[8], dg[8];
#pragma unroll
  for (int v = 0; v < 8; ++v) {
    float rm = fmaxf(fmaxf(acc[0][v], acc[1][v]), fmaxf(acc[2][v], acc[3][v]));
    rm = fmaxf(rm, __shfl_xor(rm, 1)); rm = fmaxf(rm, __shfl_xor(rm, 2));
    rm = fmaxf(rm, __shfl_xor(rm, 4)); rm = fmaxf(rm, __shfl_xor(rm, 8));
    stab[v] = rm;                               // stabilizer for row half*8+v
    dg[v] = __shfl(ss, half * 8 + v) * 0.0625f; // diag = sumsq * 0.5 * d^-0.5
  }
#pragma unroll
  for (int j = 0; j < 4; ++j) {
    int col = j * 16 + n16;
#pragma unroll
    for (int v = 0; v < 8; ++v) {
      long r = row0 + half * 8 + v;
      float val = 0.125f * __expf(acc[j][v] - dg[v] - stab[v]) + 1e-4f;  // ratio=M^-0.5
      qprime[((long)bh * NSEQ + r) * MM + col] = (__bf16)val;
    }
  }
}

// FAVOR+ for k, pass 1: global max of data_dash per (b,h)
__global__ void favorp_k_max(const __bf16* __restrict__ Kh, const __bf16* __restrict__ projsw,
                             float* __restrict__ stab) {
  int wave = threadIdx.x >> 5, lane = threadIdx.x & 31;
  int tile = blockIdx.x * (blockDim.x >> 5) + wave;
  int bh = blockIdx.y, b = bh >> 4, h = bh & 15;
  long row0 = (long)tile * 16;
  const __bf16* abase = Kh + (row0 * BB + b) * EE + h * DD;
  v8f acc[4] = {}; v16bf a0, a1;
  ddash_tile(abase, (long)BB * EE, projsw + (long)h * 8 * 512, acc, a0, a1);
  float m = -3.0e38f;
#pragma unroll
  for (int j = 0; j < 4; ++j)
#pragma unroll
    for (int v = 0; v < 8; ++v) m = fmaxf(m, acc[j][v]);
  m = fmaxf(m, __shfl_xor(m, 1)); m = fmaxf(m, __shfl_xor(m, 2));
  m = fmaxf(m, __shfl_xor(m, 4)); m = fmaxf(m, __shfl_xor(m, 8));
  m = fmaxf(m, __shfl_xor(m, 16));
  if (lane == 0) atomicMaxF(&stab[bh], m);
}

// FAVOR+ for k, pass 2: emit k' transposed -> kprimeT[bh][m][n] bf16
__global__ void favorp_k_emit(const __bf16* __restrict__ Kh, const __bf16* __restrict__ projsw,
                              const float* __restrict__ stab, __bf16* __restrict__ kprimeT) {
  int wave = threadIdx.x >> 5, lane = threadIdx.x & 31;
  int tile = blockIdx.x * (blockDim.x >> 5) + wave;
  int bh = blockIdx.y, b = bh >> 4, h = bh & 15;
  long row0 = (long)tile * 16;
  const __bf16* abase = Kh + (row0 * BB + b) * EE + h * DD;
  v8f acc[4] = {}; v16bf a0, a1;
  ddash_tile(abase, (long)BB * EE, projsw + (long)h * 8 * 512, acc, a0, a1);
  float ss = rowsumsq(a0, a1);
  float st = stab[bh];
  int half = lane >> 4, n16 = lane & 15;
  float dg[8];
#pragma unroll
  for (int v = 0; v < 8; ++v) dg[v] = __shfl(ss, half * 8 + v) * 0.0625f;
#pragma unroll
  for (int j = 0; j < 4; ++j) {
    int col = j * 16 + n16;  // m index
#pragma unroll
    for (int v = 0; v < 8; ++v) {
      long n = row0 + half * 8 + v;
      float val = 0.125f * __expf(acc[j][v] - dg[v] - st) + 1e-4f;
      kprimeT[((long)bh * MM + col) * NSEQ + n] = (__bf16)val;
    }
  }
}

// ksum[row] = sum over ncols of bf16 matrix row
__global__ void rowsum_bf16(const __bf16* __restrict__ src, float* __restrict__ dst, int ncols) {
  int row = blockIdx.x * (blockDim.x >> 5) + (threadIdx.x >> 5);
  int lane = threadIdx.x & 31;
  const __bf16* p = src + (long)row * ncols;
  float s = 0.f;
  for (int c = lane; c < ncols; c += 32) s += (float)p[c];
  s += __shfl_xor(s, 1); s += __shfl_xor(s, 2); s += __shfl_xor(s, 4);
  s += __shfl_xor(s, 8); s += __shfl_xor(s, 16);
  if (lane == 0) dst[row] = s;
}

// kv[bh][m][d] = sum_n k'T[m][n] * V[n][h*64+d] ; A = kprimeT (K=4096), B = swizzled V
__global__ void kv_gemm(const __bf16* __restrict__ kpt, const __bf16* __restrict__ vsw,
                        float* __restrict__ kv) {
  int bh = blockIdx.x, b = bh >> 4, h = bh & 15;
  int wave = threadIdx.x >> 5, lane = threadIdx.x & 31;
  const __bf16* abase = kpt + ((long)bh * MM + wave * 16) * NSEQ;
  v8f acc[4] = {};
  for (int kt = 0; kt < NSEQ / 32; ++kt) {
    const __bf16* f = vsw + (((long)b * 128 + kt) * 64 + h * 4) * 512;
    v16bf b0 = load_b_frag(f + 0 * 512);
    v16bf b1 = load_b_frag(f + 1 * 512);
    v16bf b2 = load_b_frag(f + 2 * 512);
    v16bf b3 = load_b_frag(f + 3 * 512);
    v16bf a = load_a_frag(abase + (long)kt * 32, NSEQ);
    acc[0] = WMMA_BF16(a, b0, acc[0]);
    acc[1] = WMMA_BF16(a, b1, acc[1]);
    acc[2] = WMMA_BF16(a, b2, acc[2]);
    acc[3] = WMMA_BF16(a, b3, acc[3]);
  }
  int half = lane >> 4, n16 = lane & 15;
#pragma unroll
  for (int j = 0; j < 4; ++j)
#pragma unroll
    for (int v = 0; v < 8; ++v)
      kv[((long)bh * MM + wave * 16 + half * 8 + v) * DD + j * 16 + n16] = acc[j][v];
}

// qkv + normalizer -> attn bf16 [NB][E]
__global__ void qkv_kernel(const __bf16* __restrict__ qp, const __bf16* __restrict__ kvsw,
                           const float* __restrict__ ksum, __bf16* __restrict__ attn) {
  __shared__ float sk[MM];
  int bh = blockIdx.y, b = bh >> 4, h = bh & 15;
  if (threadIdx.x < MM) sk[threadIdx.x] = ksum[bh * MM + threadIdx.x];
  __syncthreads();
  int wave = threadIdx.x >> 5, lane = threadIdx.x & 31;
  int tile = blockIdx.x * (blockDim.x >> 5) + wave;
  long row0 = (long)tile * 16;
  const __bf16* abase = qp + ((long)bh * NSEQ + row0) * MM;
  v8f acc[4] = {}; v16bf a0, a1;
  ddash_tile(abase, MM, kvsw + (long)bh * 8 * 512, acc, a0, a1);
  int half = lane >> 4, n16 = lane & 15;
  float part = 0.f;
#pragma unroll
  for (int i = 0; i < 8; ++i) {
    part += (float)a0[i]     * sk[half * 16 + i];
    part += (float)a0[8 + i] * sk[half * 16 + 8 + i];
    part += (float)a1[i]     * sk[32 + half * 16 + i];
    part += (float)a1[8 + i] * sk[32 + half * 16 + 8 + i];
  }
  part += __shfl_xor(part, 16);  // lane L: normalizer of row (L&15)
#pragma unroll
  for (int j = 0; j < 4; ++j) {
    int dcol = j * 16 + n16;
#pragma unroll
    for (int v = 0; v < 8; ++v) {
      int r = half * 8 + v;
      float nr = fmaxf(__shfl(part, r), 0.01f);
      long n = row0 + r;
      attn[(n * BB + b) * EE + h * DD + dcol] = (__bf16)(acc[j][v] / nr);
    }
  }
}

__global__ void init_stab(float* s) { if (threadIdx.x < BH) s[threadIdx.x] = -3.0e38f; }

// ---------------- host ----------------
extern "C" void kernel_launch(void* const* d_in, const int* in_sizes, int n_in,
                              void* d_out, int out_size, void* d_ws, size_t ws_size,
                              hipStream_t stream) {
  const float* q    = (const float*)d_in[0];
  const float* k    = (const float*)d_in[1];
  const float* v    = (const float*)d_in[2];
  const float* Wq   = (const float*)d_in[3];
  const float* bq   = (const float*)d_in[4];
  const float* Wk   = (const float*)d_in[5];
  const float* bk   = (const float*)d_in[6];
  const float* Wv   = (const float*)d_in[7];
  const float* bv   = (const float*)d_in[8];
  const float* Wo   = (const float*)d_in[9];
  const float* bo   = (const float*)d_in[10];
  const float* proj = (const float*)d_in[11];
  float* out = (float*)d_out;
  char* ws = (char*)d_ws;

  size_t off = 0;
  auto alloc = [&](size_t bytes) { size_t o = off; off = (off + bytes + 255) & ~(size_t)255; return o; };
  const size_t SZ_X = (size_t)NB * EE * sizeof(__bf16);
  __bf16* Xq   = (__bf16*)(ws + alloc(SZ_X));
  __bf16* Xk   = (__bf16*)(ws + alloc(SZ_X));
  __bf16* Xv   = (__bf16*)(ws + alloc(SZ_X));
  __bf16* Wqs  = (__bf16*)(ws + alloc((size_t)EE * EE * 2));
  __bf16* Wks  = (__bf16*)(ws + alloc((size_t)EE * EE * 2));
  __bf16* Wvs  = (__bf16*)(ws + alloc((size_t)EE * EE * 2));
  __bf16* Wos  = (__bf16*)(ws + alloc((size_t)EE * EE * 2));
  __bf16* Pjs  = (__bf16*)(ws + alloc((size_t)HH * 8 * 512 * 2));
  __bf16* Qh   = (__bf16*)(ws + alloc(SZ_X));
  __bf16* Kh   = (__bf16*)(ws + alloc(SZ_X));
  __bf16* Vh   = (__bf16*)(ws + alloc(SZ_X));
  __bf16* Vsw  = (__bf16*)(ws + alloc((size_t)BB * 8192 * 512 * 2));
  __bf16* Qp   = (__bf16*)(ws + alloc((size_t)BH * NSEQ * MM * 2));
  __bf16* KpT  = (__bf16*)(ws + alloc((size_t)BH * MM * NSEQ * 2));
  float*  Stab = (float*)(ws + alloc(BH * 4));
  float*  Ksum = (float*)(ws + alloc((size_t)BH * MM * 4));
  float*  Kv   = (float*)(ws + alloc((size_t)BH * MM * DD * 4));
  __bf16* Kvsw = (__bf16*)(ws + alloc((size_t)BH * 8 * 512 * 2));
  __bf16* Attn = (__bf16*)(ws + alloc(SZ_X));

  const long NELM = (long)NB * EE;
  const float data_norm = 0.35355339059327373f;  // 64^-0.25

  // 1) convert inputs to bf16 (already [NB,E] row-major with row = n*B+b)
  cvt_f32_bf16<<<NELM / 256, 256, 0, stream>>>(q, Xq, NELM);
  cvt_f32_bf16<<<NELM / 256, 256, 0, stream>>>(k, Xk, NELM);
  cvt_f32_bf16<<<NELM / 256, 256, 0, stream>>>(v, Xv, NELM);

  // 2) swizzle weights: B[k=e][n=f] = W[f,e]  (strideK=1, strideN=E)
  swizzleB<float><<<dim3(256, 1), 256, 0, stream>>>(Wq, Wqs, 0, 1, EE, 64, 2048, 1.f);
  swizzleB<float><<<dim3(256, 1), 256, 0, stream>>>(Wk, Wks, 0, 1, EE, 64, 2048, 1.f);
  swizzleB<float><<<dim3(256, 1), 256, 0, stream>>>(Wv, Wvs, 0, 1, EE, 64, 2048, 1.f);
  swizzleB<float><<<dim3(256, 1), 256, 0, stream>>>(Wo, Wos, 0, 1, EE, 64, 2048, 1.f);
  // projT per head, with data_normalizer folded in: B[k=d][n=m] = proj[h][m][d]*nrm
  swizzleB<float><<<dim3(1, HH), 256, 0, stream>>>(proj, Pjs, (long)MM * DD, 1, DD, 4, 8, data_norm);

  // 3) q/k/v linear layers (bf16 WMMA, f32 accumulate, TDM-staged B, 32x64/wave)
  gemm_bias<<<dim3(64, 16), 256, 0, stream>>>(Xq, Wqs, bq, Qh, (float*)nullptr, 32, 64, EE, EE);
  gemm_bias<<<dim3(64, 16), 256, 0, stream>>>(Xk, Wks, bk, Kh, (float*)nullptr, 32, 64, EE, EE);
  gemm_bias<<<dim3(64, 16), 256, 0, stream>>>(Xv, Wvs, bv, Vh, (float*)nullptr, 32, 64, EE, EE);

  // 4) swizzle Vh for kv GEMM: per b, B[k=n][n=f] = Vh[(n*B+b)*E + f]
  swizzleB<__bf16><<<dim3(1024, BB), 256, 0, stream>>>(Vh, Vsw, EE, (long)BB * EE, 1, 64, 8192, 1.f);

  // 5) FAVOR+
  favorp_q_kernel<<<dim3(32, BH), 256, 0, stream>>>(Qh, Pjs, Qp);
  init_stab<<<1, 64, 0, stream>>>(Stab);
  favorp_k_max<<<dim3(32, BH), 256, 0, stream>>>(Kh, Pjs, Stab);
  favorp_k_emit<<<dim3(32, BH), 256, 0, stream>>>(Kh, Pjs, Stab, KpT);
  rowsum_bf16<<<dim3(512), 256, 0, stream>>>(KpT, Ksum, NSEQ);

  // 6) kv = k'^T @ v per (b,h); then swizzle kv for qkv GEMM
  kv_gemm<<<dim3(BH), 128, 0, stream>>>(KpT, Vsw, Kv);
  swizzleB<float><<<dim3(1, BH), 256, 0, stream>>>(Kv, Kvsw, (long)MM * DD, DD, 1, 4, 8, 1.f);

  // 7) qkv + normalizer -> attn bf16
  qkv_kernel<<<dim3(32, BH), 256, 0, stream>>>(Qp, Kvsw, Ksum, Attn);

  // 8) output projection -> d_out f32 ([N,B,E] == row-major [NB,E])
  gemm_bias<<<dim3(64, 16), 256, 0, stream>>>(Attn, Wos, bo, (__bf16*)nullptr, out, 32, 64, EE, EE);
}
